// SoftMergingExpertLayer_54228257079316
// MI455X (gfx1250) — matrix operand order repaired and checked by reading, not compiled, not run on hardware
//
#include <hip/hip_runtime.h>
#include <hip/hip_bf16.h>

// ---------------------------------------------------------------------------
// SMEAR soft-merged MoE FFN as two flattened-K GEMMs (K = E*KSRC), A rows
// scaled on the fly by routing weights:
//   h = gelu( (r.X) @ W1_flat + b1_m )   M=128 N=2048 K=12288
//   y =       (r.H) @ W2_flat + b2_m    M=128 N=768  K=32768
// Memory-bound: 201 MB f32 weights read once -> ~8.6 us HBM floor @23.3TB/s.
// v_wmma_f32_16x16x32_bf16 (f32 accumulate), weights streamed into LDS with
// GLOBAL_LOAD_ASYNC_TO_LDS_B128 (ASYNCcnt), K-split across WGs for occupancy
// with deterministic per-segment partial buffers + reduction epilogues.
// ---------------------------------------------------------------------------

#define SMEAR_ASYNC_B 1   // use global_load_async_to_lds_b128 for weight tiles

typedef __bf16 bf16_t;
typedef bf16_t bf16x16 __attribute__((ext_vector_type(16)));
typedef bf16_t bf16x8  __attribute__((ext_vector_type(8)));
typedef float  f32x8   __attribute__((ext_vector_type(8)));

constexpr int BS = 128;   // batch*seq (M)
constexpr int H  = 768;   // hidden
constexpr int F  = 2048;  // ffn dim
constexpr int E  = 16;    // experts

constexpr int NT = 64;    // N columns per workgroup (4 x 16 per wave)
constexpr int KT = 64;    // K per staging step (2 WMMA k-subtiles)

constexpr int KSEG1 = 8,  EPS1 = 2;  // layer1: 8 segments x 2 experts (K=1536)
constexpr int KSEG2 = 16, EPS2 = 1;  // layer2: 16 segments x 1 expert (K=2048)

// padded LDS row strides (break 128B-stride bank conflicts)
constexpr int LDA  = KT + 8;   // 72 bf16  (144 B)
constexpr int LDB  = NT + 16;  // 80 bf16  (160 B)
constexpr int LDBF = NT + 4;   // 68 f32   (272 B)

// ------------------------- routing softmax ---------------------------------
__global__ __launch_bounds__(128) void smear_softmax_kernel(
    const float* __restrict__ logits, float* __restrict__ r)
{
    const int t = threadIdx.x;  // one row per thread
    float v[E];
    float mx = -3.0e38f;
#pragma unroll
    for (int e = 0; e < E; ++e) { v[e] = logits[t * E + e]; mx = fmaxf(mx, v[e]); }
    float s = 0.0f;
#pragma unroll
    for (int e = 0; e < E; ++e) { v[e] = __expf(v[e] - mx); s += v[e]; }
    const float inv = 1.0f / s;
#pragma unroll
    for (int e = 0; e < E; ++e) r[t * E + e] = v[e] * inv;
}

// ------------------------- fused expert GEMM -------------------------------
// Workgroup = 256 threads (8 wave32): owns a 64-wide N strip, all 128 M rows
// (wave w -> rows [16w,16w+16), 4 accumulator tiles along N), and the K range
// of EPS whole experts selected by blockIdx.y.  Partial sums -> partial[seg].
template <int KSRC, int N, int EPS>
__global__ __launch_bounds__(256) void smear_gemm_kernel(
    const float* __restrict__ Asrc,   // [BS][KSRC]
    const float* __restrict__ W,      // [E*KSRC][N]
    const float* __restrict__ r,      // [BS][E]
    float* __restrict__ partial)      // [KSEG][BS][N]
{
    __shared__ __align__(16) bf16_t ldsA[BS * LDA];   // 18.0 KB scaled-A (bf16)
    __shared__ __align__(32) bf16_t ldsB[KT * LDB];   // 10.0 KB weights  (bf16)
#if SMEAR_ASYNC_B
    __shared__ __align__(16) float  ldsBf32[KT * LDBF]; // 17.0 KB async f32 staging
#endif

    const int t    = threadIdx.x;
    const int lane = t & 31;
    const int wave = t >> 5;
    const int n0   = blockIdx.x * NT;

    // A staging: thread -> row t>>1, 32-col half (t&1)
    const int mA = t >> 1;
    const int cA = (t & 1) * 32;
    // B staging: thread -> row t>>2, 16-col quarter (t&3)  (16 f32 = 64 B)
    const int bK = t >> 2;
    const int bC = (t & 3) * 16;

    // WMMA A-fragment coords (16-bit A 16x32 ISA layout)
    const int mw = wave * 16 + (lane & 15);
    const int kb8 = (lane >> 4) * 8;   // {0,8}

    f32x8 acc[4] = {};

    const int e0 = blockIdx.y * EPS;
#pragma unroll 1
    for (int ee = 0; ee < EPS; ++ee) {
        const int   e    = e0 + ee;
        const float rm   = r[mA * E + e];                       // staging row's weight
        const float* arow = Asrc + (size_t)mA * KSRC;
        const float* wblk = W + ((size_t)e * KSRC) * N + n0;    // expert e, strip n0

#pragma unroll 1
        for (int k0 = 0; k0 < KSRC; k0 += KT) {
            const float* wrow = wblk + (size_t)(k0 + bK) * N + bC;
#if SMEAR_ASYNC_B
            // ---- async-copy B tile (64x64 f32) straight into LDS ----
            {
                const unsigned ldsoff =
                    (unsigned)(uintptr_t)&ldsBf32[bK * LDBF + bC];
                const unsigned long long ga = (unsigned long long)(uintptr_t)wrow;
                asm volatile(
                    "global_load_async_to_lds_b128 %0, %1, off\n\t"
                    "global_load_async_to_lds_b128 %0, %1, off offset:16\n\t"
                    "global_load_async_to_lds_b128 %0, %1, off offset:32\n\t"
                    "global_load_async_to_lds_b128 %0, %1, off offset:48"
                    :: "v"(ldsoff), "v"(ga) : "memory");
            }
#else
            float breg[16];
#pragma unroll
            for (int j = 0; j < 16; ++j) breg[j] = wrow[j];
#endif
            // ---- stage A tile (128x64): scale by r, f32 -> bf16 ----
            {
                const float* src = arow + k0 + cA;
                bf16x8 s0, s1, s2, s3;
#pragma unroll
                for (int j = 0; j < 8; ++j) {
                    s0[j] = (bf16_t)(rm * src[j]);
                    s1[j] = (bf16_t)(rm * src[j + 8]);
                    s2[j] = (bf16_t)(rm * src[j + 16]);
                    s3[j] = (bf16_t)(rm * src[j + 24]);
                }
                bf16_t* dst = &ldsA[mA * LDA + cA];
                *(bf16x8*)(dst + 0)  = s0;
                *(bf16x8*)(dst + 8)  = s1;
                *(bf16x8*)(dst + 16) = s2;
                *(bf16x8*)(dst + 24) = s3;
            }
            // prefetch next weight tile (global_prefetch_b8)
            if (k0 + KT < KSRC)
                __builtin_prefetch(wrow + (size_t)KT * N, 0, 0);

#if SMEAR_ASYNC_B
            asm volatile("s_wait_asynccnt 0" ::: "memory");
            __syncthreads();
            // ---- one-pass f32 -> bf16 conversion of B tile (LDS -> LDS) ----
            {
                const float* s = &ldsBf32[bK * LDBF + bC];
                bf16x8 c0, c1;
#pragma unroll
                for (int j = 0; j < 8; ++j) {
                    c0[j] = (bf16_t)s[j];
                    c1[j] = (bf16_t)s[j + 8];
                }
                bf16_t* dst = &ldsB[bK * LDB + bC];
                *(bf16x8*)(dst + 0) = c0;
                *(bf16x8*)(dst + 8) = c1;
            }
#else
            {
                bf16x8 c0, c1;
#pragma unroll
                for (int j = 0; j < 8; ++j) {
                    c0[j] = (bf16_t)breg[j];
                    c1[j] = (bf16_t)breg[j + 8];
                }
                bf16_t* dst = &ldsB[bK * LDB + bC];
                *(bf16x8*)(dst + 0) = c0;
                *(bf16x8*)(dst + 8) = c1;
            }
#endif
            __syncthreads();

            // ---- 2 k-subtiles x 4 n-subtiles = 8 WMMAs per wave ----
#pragma unroll
            for (int ks = 0; ks < KT; ks += 32) {
                bf16x16 afrag;
                {
                    const bf16_t* ap = &ldsA[mw * LDA + ks + kb8];
                    bf16x8 lo = *(const bf16x8*)(ap);        // K = ks+kb8 .. +7
                    bf16x8 hi = *(const bf16x8*)(ap + 16);   // K = ks+kb8+16 .. +23
#pragma unroll
                    for (int i = 0; i < 8; ++i) { afrag[i] = lo[i]; afrag[i + 8] = hi[i]; }
                }
#pragma unroll
                for (int ns = 0; ns < 4; ++ns) {
                    // B layout: lane = N column (lane&15), 16 K values/lane in A's K order
                    bf16x16 bfrag;
#pragma unroll
                    for (int i = 0; i < 8; ++i) {
                        bfrag[i]     = ldsB[(ks + kb8 + i) * LDB + ns * 16 + (lane & 15)];
                        bfrag[i + 8] = ldsB[(ks + kb8 + 16 + i) * LDB + ns * 16 + (lane & 15)];
                    }
                    acc[ns] = __builtin_amdgcn_wmma_f32_16x16x32_bf16(
                        false, afrag, false, bfrag, (short)0, acc[ns], false, false);
                }
            }
            __syncthreads();
        }
    }

    // ---- store partial tile (plain stores: deterministic) ----
    // C/D layout: VGPR j -> M = j + 8*(lane>>4); N = lane&15
    float* pout = partial + ((size_t)blockIdx.y * BS) * N;
    const int nl = lane & 15;
#pragma unroll
    for (int ns = 0; ns < 4; ++ns) {
        const int n = n0 + ns * 16 + nl;
#pragma unroll
        for (int j = 0; j < 8; ++j) {
            const int m = wave * 16 + j + 8 * (lane >> 4);
            pout[(size_t)m * N + n] = acc[ns][j];
        }
    }
}

// --------------- segment reduction + merged bias (+ gelu) ------------------
template <int SEG, int N, bool GELU>
__global__ __launch_bounds__(256) void smear_reduce_kernel(
    const float* __restrict__ partial,  // [SEG][BS][N]
    const float* __restrict__ bias,     // [E][N]
    const float* __restrict__ r,        // [BS][E]
    float* __restrict__ out)            // [BS][N]
{
    const int idx = blockIdx.x * 256 + threadIdx.x;
    const int m = idx / N;
    const int n = idx % N;
    float s = 0.0f;
#pragma unroll
    for (int seg = 0; seg < SEG; ++seg) s += partial[(size_t)seg * BS * N + idx];
    float b = 0.0f;
#pragma unroll
    for (int e = 0; e < E; ++e) b += r[m * E + e] * bias[e * N + n];
    float v = s + b;
    if (GELU) {
        const float c = 0.7978845608028654f * (v + 0.044715f * v * v * v);
        v = 0.5f * v * (1.0f + tanhf(c));
    }
    out[idx] = v;
}

// ---------------------------------------------------------------------------
extern "C" void kernel_launch(void* const* d_in, const int* in_sizes, int n_in,
                              void* d_out, int out_size, void* d_ws, size_t ws_size,
                              hipStream_t stream) {
    (void)in_sizes; (void)n_in; (void)out_size; (void)ws_size;

    const float* x      = (const float*)d_in[0];  // [BS][H]
    const float* logits = (const float*)d_in[1];  // [BS][E]
    const float* W1     = (const float*)d_in[2];  // [E*H][F]
    const float* b1     = (const float*)d_in[3];  // [E][F]
    const float* W2     = (const float*)d_in[4];  // [E*F][H]
    const float* b2     = (const float*)d_in[5];  // [E][H]
    float*       out    = (float*)d_out;          // [BS][H]

    float* r  = (float*)d_ws;            // [BS][E]          8 KB
    float* h  = r  + (size_t)BS * E;     // [BS][F]          1 MB
    float* p1 = h  + (size_t)BS * F;     // [KSEG1][BS][F]   8 MB
    float* p2 = p1 + (size_t)KSEG1 * BS * F;  // [KSEG2][BS][H]  6 MB

    smear_softmax_kernel<<<1, 128, 0, stream>>>(logits, r);

    smear_gemm_kernel<H, F, EPS1>
        <<<dim3(F / NT, KSEG1), 256, 0, stream>>>(x, W1, r, p1);
    smear_reduce_kernel<KSEG1, F, true>
        <<<(BS * F) / 256, 256, 0, stream>>>(p1, b1, r, h);

    smear_gemm_kernel<F, H, EPS2>
        <<<dim3(H / NT, KSEG2), 256, 0, stream>>>(h, W2, r, p2);
    smear_reduce_kernel<KSEG2, H, false>
        <<<(BS * H) / 256, 256, 0, stream>>>(p2, b2, r, out);
}
